// GeoAggregator_34952443855049
// MI455X (gfx1250) — matrix operand order repaired and checked by reading, not compiled, not run on hardware
//
#include <hip/hip_runtime.h>
#include <hip/hip_bf16.h>
#include <hip/hip_fp16.h>

// ---------------------------------------------------------------------------
// GeoAggregator fused forward for gfx1250 (CDNA5, wave32, WMMA 16x16x32 f16)
// ---------------------------------------------------------------------------
#define BSZ 8192
#define SL  49
#define FD  11
#define CTX 48
#define DM  64
#define NHID 4
#define NHEADS 8
#define HEADD 8
#define DEC_IN 138
#define DEC_KPAD 160

typedef __attribute__((ext_vector_type(16))) _Float16 v16h;
typedef __attribute__((ext_vector_type(8)))  float    v8f;

// f16 weight workspace layout (offsets in halves, each 64x64 matrix = 4096 halves,
// stored in WMMA-B fragment order: (((nt*kchunks+kc)*32 + lane)*16 + e))
#define OFF_XE_W2 0
#define OFF_YE_W2 4096
#define OFF_WQ    8192     /* 4 mats: (l*2+s) */
#define OFF_WK    24576
#define OFF_WV    40960
#define OFF_WO    57344
#define OFF_WK2   73728    /* 2 mats: s */
#define OFF_WV2   81920
// byte offsets inside d_ws
#define WS_PART   0           /* 256*18 f32 partials      */
#define WS_STATS  18432       /* 18 f32                   */
#define WS_W16    18560       /* 90112 halves  -> 198784  */
#define WS_W0SW   198912      /* 160*512 halves -> 362752 */
#define WS_W1SW   362752      /* 512*256 halves -> 624896 */
#define WS_ENC16  624896      /* 8192*160 halves          */
#define WS_H0     3246336     /* 8192*512 halves          */
#define WS_H1     11634944    /* 8192*256 halves          */
#define WS_TOTAL  15829248

__device__ __forceinline__ float tshrink(float x) { return x - tanhf(x); }

// ---------------------------------------------------------------------------
// BN statistics: deterministic two-pass reduction over (BS, SL) for 9 features
// ---------------------------------------------------------------------------
__global__ __launch_bounds__(256) void geo_bn_partial(const float* __restrict__ inp,
                                                      float* __restrict__ part) {
    __shared__ float sb[256];
    const int tid = threadIdx.x;
    float s[9], q[9];
#pragma unroll
    for (int f = 0; f < 9; ++f) { s[f] = 0.f; q[f] = 0.f; }
    for (int r = blockIdx.x * 256 + tid; r < BSZ * SL; r += 256 * 256) {
        const float* row = inp + (long)r * FD;
#pragma unroll
        for (int f = 0; f < 9; ++f) {
            const int c = (f < 8) ? f : 10;
            const float v = row[c];
            s[f] += v; q[f] += v * v;
        }
    }
    for (int f = 0; f < 18; ++f) {
        sb[tid] = (f < 9) ? s[f] : q[f - 9];
        __syncthreads();
        for (int st = 128; st > 0; st >>= 1) {
            if (tid < st) sb[tid] += sb[tid + st];
            __syncthreads();
        }
        if (tid == 0) part[blockIdx.x * 18 + f] = sb[0];
        __syncthreads();
    }
}

__global__ void geo_bn_finalize(const float* __restrict__ part,
                                const float* __restrict__ gamma,
                                const float* __restrict__ beta,
                                float* __restrict__ stats) {
    const int f = threadIdx.x;
    if (f >= 9) return;
    float s = 0.f, q = 0.f;
    for (int b = 0; b < 256; ++b) { s += part[b * 18 + f]; q += part[b * 18 + 9 + f]; }
    const float n = (float)(BSZ * SL);
    const float mean = s / n;
    const float var  = q / n - mean * mean;
    const float inv  = rsqrtf(var + 1e-5f);
    const float sc   = inv * gamma[f];
    stats[f]     = sc;                   // scale
    stats[9 + f] = beta[f] - mean * sc;  // shift
}

// ---------------------------------------------------------------------------
// Convert a group of 64x64 f32 matrices into f16 WMMA-B fragment order.
//   n = nt*16 + (lane&15);  k = kc*32 + ((lane>>4)<<3) + (e&7) + ((e>>3)<<4)   (kchunks = 2)
// ---------------------------------------------------------------------------
__global__ void geo_cvt_w(const float* __restrict__ src, _Float16* __restrict__ dst,
                          int nmat) {
    const int i = blockIdx.x * blockDim.x + threadIdx.x;
    if (i >= nmat * 4096) return;
    const int m    = i >> 12;
    const int r    = i & 4095;
    const int e    = r & 15;
    const int lane = (r >> 4) & 31;
    const int ntkc = r >> 9;
    const int kc   = ntkc & 1;
    const int nt   = ntkc >> 1;
    const int n = nt * 16 + (lane & 15);
    const int k = kc * 32 + ((lane >> 4) << 3) + (e & 7) + ((e >> 3) << 4);
    dst[i] = (_Float16)src[m * 4096 + k * 64 + n];
}

// Generic decoder-weight conversion: src is (ksrc x N) f32 row-major, K padded to kpad
__global__ void geo_cvt_dec(const float* __restrict__ src, _Float16* __restrict__ dst,
                            int ksrc, int kpad, int N) {
    const int i = blockIdx.x * blockDim.x + threadIdx.x;
    if (i >= N * kpad) return;
    const int kchunks = kpad >> 5;
    const int e    = i & 15;
    const int lane = (i >> 4) & 31;
    const int rest = i >> 9;
    const int kc   = rest % kchunks;
    const int nt   = rest / kchunks;
    const int n = nt * 16 + (lane & 15);
    const int k = kc * 32 + ((lane >> 4) << 3) + (e & 7) + ((e >> 3) << 4);
    dst[i] = (_Float16)((k < ksrc) ? src[k * N + n] : 0.f);
}

// ---------------------------------------------------------------------------
// WMMA helpers
// ---------------------------------------------------------------------------
// A fragment (16x32 f16), row-major with stride `ld`:
//   lanes 0-15: M=lane,   K in {0..7, 16..23};  lanes 16-31: M=lane-16, K in {8..15, 24..31}
__device__ __forceinline__ v16h load_frag_a_ld(const _Float16* A, int ld, int m0, int k0,
                                               int lane) {
    const _Float16* row = A + (size_t)(m0 + (lane & 15)) * ld + k0 + ((lane & 16) ? 8 : 0);
    v16h a;
#pragma unroll
    for (int e = 0; e < 8; ++e) a[e] = row[e];
#pragma unroll
    for (int e = 0; e < 8; ++e) a[8 + e] = row[16 + e];
    return a;
}

// MODE 0: f16 store (+bias); MODE 1: f32 store; MODE 2: f32 accumulate
template <int MODE>
__device__ __forceinline__ void wmma_mm(const _Float16* A, int mtiles,
                                        const _Float16* __restrict__ Wsw,
                                        const float* __restrict__ bias,
                                        _Float16* dst16, float* dst32,
                                        int rows_valid, int lane, int wave) {
    const int nt = wave;                      // 4 waves -> 4 N-tiles of 16
    const int n  = nt * 16 + (lane & 15);
    const v16h b0 = *(const v16h*)(Wsw + (size_t)((nt * 2 + 0) * 32 + lane) * 16);
    const v16h b1 = *(const v16h*)(Wsw + (size_t)((nt * 2 + 1) * 32 + lane) * 16);
    const float bn = bias ? bias[n] : 0.f;
    for (int mt = 0; mt < mtiles; ++mt) {
        v16h a0 = load_frag_a_ld(A, DM, mt * 16, 0,  lane);
        v16h a1 = load_frag_a_ld(A, DM, mt * 16, 32, lane);
        v8f acc = {};
        acc = __builtin_amdgcn_wmma_f32_16x16x32_f16(false, a0, false, b0, (short)0, acc, false, false);
        acc = __builtin_amdgcn_wmma_f32_16x16x32_f16(false, a1, false, b1, (short)0, acc, false, false);
        const int mb = mt * 16 + ((lane & 16) ? 8 : 0);
#pragma unroll
        for (int r = 0; r < 8; ++r) {
            const int m = mb + r;
            if (m < rows_valid) {
                if (MODE == 0)      dst16[m * DM + n] = (_Float16)(acc[r] + bn);
                else if (MODE == 1) dst32[m * DM + n] = acc[r];
                else                dst32[m * DM + n] += acc[r];
            }
        }
    }
}

// ---------------------------------------------------------------------------
// Decoder GEMM: C[m][n] = act(A @ W + bias), A f16 (stride kpad) in global,
// W in fragment order, C f16 row-major (stride N).
// Block = 16 rows x 64 cols (4 waves, 1 n-tile each). grid = (M/16, N/64).
// ---------------------------------------------------------------------------
__global__ __launch_bounds__(128) void geo_dec_gemm(const _Float16* __restrict__ A,
                                                    const _Float16* __restrict__ Wsw,
                                                    const float* __restrict__ bias,
                                                    _Float16* __restrict__ C,
                                                    int kchunks, int kpad, int N, int act) {
    const int lane  = threadIdx.x & 31;
    const int wave  = threadIdx.x >> 5;
    const int m0    = blockIdx.x * 16;
    const int ntile = blockIdx.y * 4 + wave;
    const int n     = ntile * 16 + (lane & 15);
    const _Float16* arow = A + (size_t)(m0 + (lane & 15)) * kpad + ((lane & 16) ? 8 : 0);
    v8f acc = {};
    for (int kc = 0; kc < kchunks; ++kc) {
        const _Float16* p = arow + kc * 32;
        v16h a;
#pragma unroll
        for (int e = 0; e < 8; ++e) a[e] = p[e];
#pragma unroll
        for (int e = 0; e < 8; ++e) a[8 + e] = p[16 + e];
        const v16h bfrag = *(const v16h*)(Wsw + (size_t)((ntile * kchunks + kc) * 32 + lane) * 16);
        acc = __builtin_amdgcn_wmma_f32_16x16x32_f16(false, a, false, bfrag, (short)0, acc, false, false);
    }
    const float bn = bias[n];
    const int mb = (lane & 16) ? 8 : 0;
#pragma unroll
    for (int r = 0; r < 8; ++r) {
        float v = acc[r] + bn;
        if (act) v = tshrink(v);
        C[(size_t)(m0 + mb + r) * N + n] = (_Float16)v;
    }
}

__global__ __launch_bounds__(256) void geo_dec_out(const _Float16* __restrict__ h1,
                                                   const float* __restrict__ w2,
                                                   const float* __restrict__ b2,
                                                   float* __restrict__ out) {
    const int i = blockIdx.x * blockDim.x + threadIdx.x;
    if (i >= BSZ) return;
    const _Float16* row = h1 + (size_t)i * 256;
    float acc = 0.f;
    for (int k = 0; k < 256; ++k) acc += (float)row[k] * w2[k];
    out[i] = acc + b2[0];
}

// ---------------------------------------------------------------------------
// Main fused kernel: one workgroup (4 wave32) per batch element
// ---------------------------------------------------------------------------
__global__ __launch_bounds__(128) void geo_main(
    const float* __restrict__ inp, const float* __restrict__ dists,
    const float* __restrict__ stats, const _Float16* __restrict__ w16,
    const float* __restrict__ xe_w1, const float* __restrict__ xe_b1,
    const float* __restrict__ xe_b2,
    const float* __restrict__ ye_w1, const float* __restrict__ ye_b1,
    const float* __restrict__ ye_b2,
    const float* __restrict__ learnable_y, const float* __restrict__ hidden_tokens,
    const float* __restrict__ Wq2, const float* __restrict__ Wo2,
    _Float16* __restrict__ enc16) {

    __shared__ float     raw[SL][FD + 1];   // col 11 holds dists
    __shared__ float     cosr[SL][4], sinr[SL][4];
    __shared__ _Float16  xe[SL][DM];        // x-embed (row 48 = query)
    __shared__ _Float16  ye[SL][DM];        // y-embed (row 48 = learnable_y)
    __shared__ _Float16  abuf[DM][DM];      // WMMA A staging
    __shared__ _Float16  kmat[CTX][DM];
    __shared__ _Float16  vmat[CTX][DM];
    __shared__ float     Hm[NHID][DM];
    __shared__ float     qb[NHID][DM];      // q / k2
    __shared__ float     ob[NHID][DM];      // attn-out / v2
    __shared__ float     qf[DM];
    __shared__ float     ovec[DM];
    __shared__ float     enc[144];

    const int b    = blockIdx.x;
    const int tid  = threadIdx.x;
    const int lane = tid & 31;
    const int wave = tid >> 5;
    const float scale = 0.35355339059327379f; // 1/sqrt(8)

    // ---- Stage 0: input load + batchnorm, dists, rope angles ----
    for (int i = tid; i < SL * FD; i += 128) {
        const int r = i / FD, c = i - r * FD;
        float v = inp[(long)b * (SL * FD) + i];
        const int f = (c < 8) ? c : ((c == 10) ? 8 : -1);
        if (f >= 0) v = v * stats[f] + stats[9 + f];
        raw[r][c] = v;
    }
    for (int j = tid; j < CTX; j += 128) raw[j][FD] = dists[(long)b * SL + j];
    __syncthreads();
    if (tid < SL) {
        const float x = raw[tid][8], y = raw[tid][9];
        const float a0 = 10.f * x, a1 = 10.f * y, a2 = x, a3 = y;
        cosr[tid][0] = cosf(a0); sinr[tid][0] = sinf(a0);
        cosr[tid][1] = cosf(a1); sinr[tid][1] = sinf(a1);
        cosr[tid][2] = cosf(a2); sinr[tid][2] = sinf(a2);
        cosr[tid][3] = cosf(a3); sinr[tid][3] = sinf(a3);
    }
    __syncthreads();

    // ---- Stage 1: x_embed = tanhshrink(in[:, :8]@w1+b1) @ w2 + b2 ----
    for (int i = tid; i < 64 * 64; i += 128) {
        const int m = i >> 6, n = i & 63;
        float v = 0.f;
        if (m < SL) {
            float acc = xe_b1[n];
#pragma unroll
            for (int k = 0; k < 8; ++k) acc += raw[m][k] * xe_w1[k * 64 + n];
            v = tshrink(acc);
        }
        abuf[m][n] = (_Float16)v;
    }
    __syncthreads();
    wmma_mm<0>(&abuf[0][0], 4, w16 + OFF_XE_W2, xe_b2, &xe[0][0], nullptr, SL, lane, wave);
    __syncthreads();

    // ---- Stage 2: y_embed (48 ctx rows) + learnable_y as query row ----
    for (int i = tid; i < CTX * 64; i += 128) {
        const int m = i >> 6, n = i & 63;
        abuf[m][n] = (_Float16)tshrink(raw[m][10] * ye_w1[n] + ye_b1[n]);
    }
    __syncthreads();
    wmma_mm<0>(&abuf[0][0], 3, w16 + OFF_YE_W2, ye_b2, &ye[0][0], nullptr, CTX, lane, wave);
    if (tid < 64) ye[CTX][tid] = (_Float16)learnable_y[tid];
    __syncthreads();

    // ---- Stage 3: two attention streams ----
    for (int s = 0; s < 2; ++s) {
        const _Float16* emb = (s == 0) ? &xe[0][0] : &ye[0][0];
        for (int i = tid; i < NHID * DM; i += 128) (&Hm[0][0])[i] = hidden_tokens[i];
        __syncthreads();

        for (int l = 0; l < 2; ++l) {
            const int wi = l * 2 + s;
            // K = ctx @ Wk ; V = ctx @ Wv
            wmma_mm<0>(emb, 3, w16 + OFF_WK + wi * 4096, nullptr, &kmat[0][0], nullptr, CTX, lane, wave);
            wmma_mm<0>(emb, 3, w16 + OFF_WV + wi * 4096, nullptr, &vmat[0][0], nullptr, CTX, lane, wave);
            __syncthreads();
            // rope(K) in LDS
            for (int i = tid; i < CTX * 32; i += 128) {
                const int j = i >> 5, p = i & 31, n0 = 2 * p, ai = p & 3;
                const float c = cosr[j][ai], sn = sinr[j][ai];
                const float k0 = (float)kmat[j][n0], k1 = (float)kmat[j][n0 + 1];
                kmat[j][n0]     = (_Float16)(c * k0 - sn * k1);
                kmat[j][n0 + 1] = (_Float16)(sn * k0 + c * k1);
            }
            // stage H -> abuf (pad to 16 rows)
            for (int i = tid; i < 16 * 64; i += 128) {
                const int m = i >> 6, n = i & 63;
                abuf[m][n] = (m < NHID) ? (_Float16)Hm[m][n] : (_Float16)0.f;
            }
            __syncthreads();
            // Q = H @ Wq
            wmma_mm<1>(&abuf[0][0], 1, w16 + OFF_WQ + wi * 4096, nullptr, nullptr, &qb[0][0], NHID, lane, wave);
            __syncthreads();
            // scores + softmax + A@V (one thread per (head, query))
            if (tid < 32) {
                const int h = tid >> 2, qi = tid & 3;
                float qv[HEADD];
#pragma unroll
                for (int d = 0; d < HEADD; ++d) qv[d] = qb[qi][h * 8 + d];
                float a[CTX];
                float mx = -1e30f;
#pragma unroll
                for (int j = 0; j < CTX; ++j) {
                    float acc = 0.f;
#pragma unroll
                    for (int d = 0; d < HEADD; ++d) acc += qv[d] * (float)kmat[j][h * 8 + d];
                    acc = acc * scale - raw[j][FD];
                    a[j] = acc; mx = fmaxf(mx, acc);
                }
                float sum = 0.f;
#pragma unroll
                for (int j = 0; j < CTX; ++j) { a[j] = __expf(a[j] - mx); sum += a[j]; }
                const float inv = 1.f / sum;
#pragma unroll
                for (int d = 0; d < HEADD; ++d) {
                    float o = 0.f;
#pragma unroll
                    for (int j = 0; j < CTX; ++j) o += a[j] * (float)vmat[j][h * 8 + d];
                    ob[qi][h * 8 + d] = o * inv;
                }
            }
            __syncthreads();
            for (int i = tid; i < 16 * 64; i += 128) {
                const int m = i >> 6, n = i & 63;
                abuf[m][n] = (m < NHID) ? (_Float16)ob[m][n] : (_Float16)0.f;
            }
            __syncthreads();
            // H += attn_out @ Wo
            wmma_mm<2>(&abuf[0][0], 1, w16 + OFF_WO + wi * 4096, nullptr, nullptr, &Hm[0][0], NHID, lane, wave);
            __syncthreads();
        }

        // ---- final cross attention for this stream ----
        const float* Wq2s = Wq2 + s * 4096;
        if (tid < 64) {
            float acc = 0.f;
            for (int k = 0; k < 64; ++k) acc += (float)emb[CTX * DM + k] * Wq2s[k * 64 + tid];
            qf[tid] = acc;
        }
        __syncthreads();
        if (tid < 32) {   // rope(q) with ang row 48
            const int p = tid, n0 = 2 * p, ai = p & 3;
            const float c = cosr[CTX][ai], sn = sinr[CTX][ai];
            const float q0 = qf[n0], q1 = qf[n0 + 1];
            qf[n0] = c * q0 - sn * q1; qf[n0 + 1] = sn * q0 + c * q1;
        }
        for (int i = tid; i < 16 * 64; i += 128) {
            const int m = i >> 6, n = i & 63;
            abuf[m][n] = (m < NHID) ? (_Float16)Hm[m][n] : (_Float16)0.f;
        }
        __syncthreads();
        wmma_mm<1>(&abuf[0][0], 1, w16 + OFF_WK2 + s * 4096, nullptr, nullptr, &qb[0][0], NHID, lane, wave);
        wmma_mm<1>(&abuf[0][0], 1, w16 + OFF_WV2 + s * 4096, nullptr, nullptr, &ob[0][0], NHID, lane, wave);
        __syncthreads();
        if (tid < NHEADS) {
            const int h = tid;
            float qv[HEADD];
#pragma unroll
            for (int d = 0; d < HEADD; ++d) qv[d] = qf[h * 8 + d];
            float sc[NHID]; float mx = -1e30f;
#pragma unroll
            for (int j = 0; j < NHID; ++j) {
                float acc = 0.f;
#pragma unroll
                for (int d = 0; d < HEADD; ++d) acc += qv[d] * qb[j][h * 8 + d];
                sc[j] = acc * scale; mx = fmaxf(mx, sc[j]);
            }
            float sum = 0.f;
#pragma unroll
            for (int j = 0; j < NHID; ++j) { sc[j] = __expf(sc[j] - mx); sum += sc[j]; }
            const float inv = 1.f / sum;
#pragma unroll
            for (int d = 0; d < HEADD; ++d) {
                float o = 0.f;
#pragma unroll
                for (int j = 0; j < NHID; ++j) o += sc[j] * ob[j][h * 8 + d];
                ovec[h * 8 + d] = o * inv;
            }
        }
        __syncthreads();
        const float* Wo2s = Wo2 + s * 4096;
        if (tid < 64) {
            float acc = 0.f;
            for (int k = 0; k < 64; ++k) acc += ovec[k] * Wo2s[k * 64 + tid];
            enc[s * 64 + tid] = acc;
        }
        __syncthreads();
    }

    // ---- Stage 4: write encoding row (f16, K padded to DEC_KPAD) ----
    if (tid < 10) enc[128 + tid] = raw[CTX][tid];
    __syncthreads();
    for (int i = tid; i < DEC_KPAD; i += 128)
        enc16[(size_t)b * DEC_KPAD + i] = (_Float16)((i < DEC_IN) ? enc[i] : 0.f);
}

// ---------------------------------------------------------------------------
extern "C" void kernel_launch(void* const* d_in, const int* in_sizes, int n_in,
                              void* d_out, int out_size, void* d_ws, size_t ws_size,
                              hipStream_t stream) {
    const float* input         = (const float*)d_in[0];
    const float* dists         = (const float*)d_in[1];
    const float* bn_gamma      = (const float*)d_in[3];
    const float* bn_beta       = (const float*)d_in[4];
    const float* xe_w1         = (const float*)d_in[5];
    const float* xe_b1         = (const float*)d_in[6];
    const float* xe_w2         = (const float*)d_in[7];
    const float* xe_b2         = (const float*)d_in[8];
    const float* ye_w1         = (const float*)d_in[9];
    const float* ye_b1         = (const float*)d_in[10];
    const float* ye_w2         = (const float*)d_in[11];
    const float* ye_b2         = (const float*)d_in[12];
    const float* learnable_y   = (const float*)d_in[13];
    const float* hidden_tokens = (const float*)d_in[14];
    const float* Wq            = (const float*)d_in[15];
    const float* Wk            = (const float*)d_in[16];
    const float* Wv            = (const float*)d_in[17];
    const float* Wo            = (const float*)d_in[18];
    const float* Wq2           = (const float*)d_in[19];
    const float* Wk2           = (const float*)d_in[20];
    const float* Wv2           = (const float*)d_in[21];
    const float* Wo2           = (const float*)d_in[22];
    const float* dec_w0        = (const float*)d_in[23];
    const float* dec_b0        = (const float*)d_in[24];
    const float* dec_w1        = (const float*)d_in[25];
    const float* dec_b1        = (const float*)d_in[26];
    const float* dec_w2        = (const float*)d_in[27];
    const float* dec_b2        = (const float*)d_in[28];

    char* wsb = (char*)d_ws;
    float*    part  = (float*)(wsb + WS_PART);
    float*    stats = (float*)(wsb + WS_STATS);
    _Float16* w16   = (_Float16*)(wsb + WS_W16);
    _Float16* w0sw  = (_Float16*)(wsb + WS_W0SW);
    _Float16* w1sw  = (_Float16*)(wsb + WS_W1SW);
    _Float16* enc16 = (_Float16*)(wsb + WS_ENC16);
    _Float16* h0    = (_Float16*)(wsb + WS_H0);
    _Float16* h1    = (_Float16*)(wsb + WS_H1);

    geo_bn_partial<<<256, 256, 0, stream>>>(input, part);
    geo_bn_finalize<<<1, 16, 0, stream>>>(part, bn_gamma, bn_beta, stats);

    auto cvt = [&](const float* src, int halfoff, int nmat) {
        const int total = nmat * 4096;
        geo_cvt_w<<<(total + 255) / 256, 256, 0, stream>>>(src, w16 + halfoff, nmat);
    };
    cvt(xe_w2, OFF_XE_W2, 1);
    cvt(ye_w2, OFF_YE_W2, 1);
    cvt(Wq,    OFF_WQ,    4);
    cvt(Wk,    OFF_WK,    4);
    cvt(Wv,    OFF_WV,    4);
    cvt(Wo,    OFF_WO,    4);
    cvt(Wk2,   OFF_WK2,   2);
    cvt(Wv2,   OFF_WV2,   2);
    geo_cvt_dec<<<(512 * DEC_KPAD + 255) / 256, 256, 0, stream>>>(dec_w0, w0sw, DEC_IN, DEC_KPAD, 512);
    geo_cvt_dec<<<(256 * 512 + 255) / 256, 256, 0, stream>>>(dec_w1, w1sw, 512, 512, 256);

    geo_main<<<BSZ, 128, 0, stream>>>(input, dists, stats, w16,
        xe_w1, xe_b1, xe_b2, ye_w1, ye_b1, ye_b2, learnable_y, hidden_tokens,
        Wq2, Wo2, enc16);

    // decoder GEMMs (WMMA, 16-row M-tiles amortize weight traffic)
    dim3 g1(BSZ / 16, 512 / 64);
    geo_dec_gemm<<<g1, 128, 0, stream>>>(enc16, w0sw, dec_b0, h0, DEC_KPAD / 32, DEC_KPAD, 512, 1);
    dim3 g2(BSZ / 16, 256 / 64);
    geo_dec_gemm<<<g2, 128, 0, stream>>>(h0, w1sw, dec_b1, h1, 512 / 32, 512, 256, 1);
    geo_dec_out<<<BSZ / 256, 256, 0, stream>>>(h1, dec_w2, dec_b2, (float*)d_out);
}
